// GAT_83872121356314
// MI455X (gfx1250) — matrix operand order repaired and checked by reading, not compiled, not run on hardware
//
#include <hip/hip_runtime.h>
#include <math.h>

// ---------------- types for WMMA bf16 fragments ----------------
typedef __bf16 bf16;
typedef __attribute__((ext_vector_type(16))) __bf16 v16bf;
typedef __attribute__((ext_vector_type(8)))  __bf16 v8bf;
typedef __attribute__((ext_vector_type(8)))  float  v8f;

union Frag {
    v16bf v;
    v8bf  h[2];
};

// ---------------- float <-> ordered-uint for atomic max ----------------
__device__ __forceinline__ unsigned f2ord(float f) {
    unsigned u = __float_as_uint(f);
    return (u & 0x80000000u) ? ~u : (u | 0x80000000u);
}
__device__ __forceinline__ float ord2f(unsigned u) {
    u = (u & 0x80000000u) ? (u & 0x7FFFFFFFu) : ~u;
    return __uint_as_float(u);
}

// ---------------- conversion: f32 -> split bf16 (hi + lo) ----------------
__global__ void k_cvt_split(const float* __restrict__ x, bf16* __restrict__ hi,
                            bf16* __restrict__ lo, int n) {
    int i = blockIdx.x * blockDim.x + threadIdx.x;
    if (i >= n) return;
    float v = x[i];
    bf16 h = (bf16)v;
    hi[i] = h;
    lo[i] = (bf16)(v - (float)h);
}

// W [K, NC] row-major  ->  hiT/loT [NC, K] row-major (transposed, split)
__global__ void k_cvt_split_t(const float* __restrict__ W, bf16* __restrict__ hiT,
                              bf16* __restrict__ loT, int K, int NC) {
    int i = blockIdx.x * blockDim.x + threadIdx.x;
    if (i >= K * NC) return;
    int k = i / NC, n = i % NC;
    float v = W[i];
    bf16 h = (bf16)v;
    hiT[(size_t)n * K + k] = h;
    loT[(size_t)n * K + k] = (bf16)(v - (float)h);
}

// ---------------- WMMA GEMM: C[M,NC] = A[M,K] * B[K,NC] ----------------
// Split-bf16 (hi+lo) error-compensated: acc = ahi*bhi + ahi*blo + alo*bhi,
// f32 accumulate. Each wave owns a 16(M) x 64(N) tile -> 4 independent
// accumulator chains (hides WMMA RAW latency) and 4x reuse of A fragments.
__global__ void __launch_bounds__(256)
k_gemm_bf16(const bf16* __restrict__ Ah, const bf16* __restrict__ Al,
            const bf16* __restrict__ Bh, const bf16* __restrict__ Bl,
            float* __restrict__ Cmat, int M, int K, int NC) {
    int wid    = blockIdx.x * (blockDim.x >> 5) + (threadIdx.x >> 5);
    int tilesN = NC >> 6;                 // 64-wide N tiles
    int tilesM = (M + 15) >> 4;
    if (wid >= tilesM * tilesN) return;   // wave-uniform: EXEC stays all-ones
    int tm   = (wid / tilesN) << 4;
    int tn   = (wid % tilesN) << 6;
    int lane = threadIdx.x & 31;
    int half = lane >> 4;                 // lane group 0..15 vs 16..31
    int lm   = lane & 15;

    const bf16* aH  = Ah + (size_t)(tm + lm) * K;
    const bf16* aL  = Al + (size_t)(tm + lm) * K;
    const bf16* bH0 = Bh + (size_t)(tn + lm) * K;
    const bf16* bL0 = Bl + (size_t)(tn + lm) * K;

    v8f acc[4] = {};
    for (int k = 0; k < K; k += 32) {
        Frag fah, fal;
        // A 16x32 bf16 layout: elems 0..7 -> k = half*8..+7 ; elems 8..15 -> 16+half*8..+7
        fah.h[0] = *(const v8bf*)(aH + k + half * 8);
        fah.h[1] = *(const v8bf*)(aH + k + 16 + half * 8);
        fal.h[0] = *(const v8bf*)(aL + k + half * 8);
        fal.h[1] = *(const v8bf*)(aL + k + 16 + half * 8);
#pragma unroll
        for (int t = 0; t < 4; ++t) {
            const bf16* bh = bH0 + (size_t)(t * 16) * K;
            const bf16* bl = bL0 + (size_t)(t * 16) * K;
            Frag fbh, fbl;
            // B 32x16 bf16 layout: elems 0..7 -> k = half*8..+7 ; elems 8..15 -> 16+half*8..+7
            fbh.h[0] = *(const v8bf*)(bh + k + half * 8);
            fbh.h[1] = *(const v8bf*)(bh + k + 16 + half * 8);
            fbl.h[0] = *(const v8bf*)(bl + k + half * 8);
            fbl.h[1] = *(const v8bf*)(bl + k + 16 + half * 8);
            acc[t] = __builtin_amdgcn_wmma_f32_16x16x32_bf16(false, fah.v, false, fbh.v,
                                                             (short)0, acc[t], false, false);
            acc[t] = __builtin_amdgcn_wmma_f32_16x16x32_bf16(false, fah.v, false, fbl.v,
                                                             (short)0, acc[t], false, false);
            acc[t] = __builtin_amdgcn_wmma_f32_16x16x32_bf16(false, fal.v, false, fbh.v,
                                                             (short)0, acc[t], false, false);
        }
    }
#pragma unroll
    for (int t = 0; t < 4; ++t)
#pragma unroll
        for (int r = 0; r < 8; ++r)   // C/D layout: VGPR r -> M = r + 8*half, N = lm
            Cmat[(size_t)(tm + r + half * 8) * NC + tn + t * 16 + lm] = acc[t][r];
}

// ---------------- attention logits: al_s[n,h], al_d[n,h] ----------------
__global__ void k_attn(const float* __restrict__ Hb, const float* __restrict__ asrc,
                       const float* __restrict__ adst, float* __restrict__ als,
                       float* __restrict__ ald, int N, int H, int C) {
    int id = blockIdx.x * blockDim.x + threadIdx.x;
    if (id >= N * H) return;
    int n = id / H, h = id % H;
    const float* row = Hb + (size_t)n * H * C + (size_t)h * C;
    float ss = 0.f, sd = 0.f;
    for (int c = 0; c < C; ++c) {
        float v = row[c];
        ss += v * asrc[h * C + c];
        sd += v * adst[h * C + c];
    }
    als[id] = ss;
    ald[id] = sd;
}

// ---------------- zero-init kernels ----------------
__global__ void k_zero_f(float* p, int n) {
    int i = blockIdx.x * blockDim.x + threadIdx.x;
    if (i < n) p[i] = 0.f;
}
__global__ void k_zero_u(unsigned* p, int n) {
    int i = blockIdx.x * blockDim.x + threadIdx.x;
    if (i < n) p[i] = 0u;
}

// ---------------- edge pass 1: segment max of leaky_relu logits ----------------
__global__ void k_edge_max(const long long* __restrict__ ei, const float* __restrict__ als,
                           const float* __restrict__ ald, unsigned* __restrict__ mu,
                           int E, int N, int H) {
    int e = blockIdx.x * blockDim.x + threadIdx.x;
    if (e >= E + N) return;
    int s, d;
    if (e < E) { s = (int)ei[e]; d = (int)ei[(size_t)E + e]; }
    else       { s = d = e - E; }                // self loops appended
    for (int h = 0; h < H; ++h) {
        float v = als[s * H + h] + ald[d * H + h];
        v = v > 0.f ? v : 0.2f * v;              // leaky_relu, slope 0.2
        atomicMax(&mu[d * H + h], f2ord(v));
    }
}

// ---------------- edge pass 2: ex = exp(e - max); segment sum ----------------
__global__ void k_edge_exp(const long long* __restrict__ ei, const float* __restrict__ als,
                           const float* __restrict__ ald, const unsigned* __restrict__ mu,
                           float* __restrict__ den, float* __restrict__ ex,
                           int E, int N, int H) {
    int e = blockIdx.x * blockDim.x + threadIdx.x;
    if (e >= E + N) return;
    int s, d;
    if (e < E) { s = (int)ei[e]; d = (int)ei[(size_t)E + e]; }
    else       { s = d = e - E; }
    for (int h = 0; h < H; ++h) {
        float v = als[s * H + h] + ald[d * H + h];
        v = v > 0.f ? v : 0.2f * v;
        float m  = ord2f(mu[d * H + h]);
        float xv = expf(v - m);
        ex[(size_t)e * H + h] = xv;
        atomicAdd(&den[d * H + h], xv);
    }
}

// ---------------- edge pass 3: scatter alpha * h[src] into agg[dst] ----------------
// one block (C threads) per edge; loop over heads
__global__ void k_edge_scatter(const long long* __restrict__ ei, const float* __restrict__ ex,
                               const float* __restrict__ den, const float* __restrict__ Hb,
                               float* __restrict__ agg, int E, int N, int H, int C) {
    int e = blockIdx.x;
    int c = threadIdx.x;
    int s, d;
    if (e < E) { s = (int)ei[e]; d = (int)ei[(size_t)E + e]; }
    else       { s = d = e - E; }
    int HC = H * C;
    for (int h = 0; h < H; ++h) {
        float alpha = ex[(size_t)e * H + h] / den[d * H + h];
        atomicAdd(&agg[(size_t)d * HC + h * C + c],
                  alpha * Hb[(size_t)s * HC + h * C + c]);
    }
}

// ---------------- bias + (optional) ELU ----------------
__global__ void k_finish(const float* __restrict__ agg, const float* __restrict__ bias,
                         float* __restrict__ out, int total, int HC, int elu) {
    int i = blockIdx.x * blockDim.x + threadIdx.x;
    if (i >= total) return;
    float v = agg[i] + bias[i % HC];
    if (elu) v = v > 0.f ? v : expm1f(v);
    out[i] = v;
}

// ======================================================================
extern "C" void kernel_launch(void* const* d_in, const int* in_sizes, int n_in,
                              void* d_out, int out_size, void* d_ws, size_t ws_size,
                              hipStream_t stream) {
    const float*     x   = (const float*)d_in[0];
    const long long* ei  = (const long long*)d_in[1];
    const float*     W1  = (const float*)d_in[2];
    const float*     as1 = (const float*)d_in[3];
    const float*     ad1 = (const float*)d_in[4];
    const float*     b1  = (const float*)d_in[5];
    const float*     W2  = (const float*)d_in[6];
    const float*     as2 = (const float*)d_in[7];
    const float*     ad2 = (const float*)d_in[8];
    const float*     b2  = (const float*)d_in[9];
    const float*     W3  = (const float*)d_in[10];
    const float*     as3 = (const float*)d_in[11];
    const float*     ad3 = (const float*)d_in[12];
    const float*     b3  = (const float*)d_in[13];

    const int HC   = in_sizes[5];           // 512
    const int Din  = in_sizes[2] / HC;      // 256
    const int N    = in_sizes[0] / Din;     // 10000
    const int E    = in_sizes[1] / 2;       // 320000
    const int H    = 4;
    const int C    = HC / H;                // 128
    const int Dout = in_sizes[13];          // 64
    const int Etot = E + N;
    const int maxK = (Din > HC) ? Din : HC; // 512

    // ---- carve workspace ----
    char*  wp  = (char*)d_ws;
    size_t off = 0;
    auto carve = [&](size_t bytes) -> void* {
        void* p = wp + off;
        off += (bytes + 255) & ~(size_t)255;
        return p;
    };
    bf16*     XBh = (bf16*)carve((size_t)N * maxK * sizeof(bf16));
    bf16*     XBl = (bf16*)carve((size_t)N * maxK * sizeof(bf16));
    bf16*     WBh = (bf16*)carve((size_t)maxK * HC * sizeof(bf16));
    bf16*     WBl = (bf16*)carve((size_t)maxK * HC * sizeof(bf16));
    float*    HB  = (float*)carve((size_t)N * HC * sizeof(float));
    float*    AGG = (float*)carve((size_t)N * HC * sizeof(float));
    float*    XF  = (float*)carve((size_t)N * HC * sizeof(float));
    float*    ALS = (float*)carve((size_t)N * H * sizeof(float));
    float*    ALD = (float*)carve((size_t)N * H * sizeof(float));
    unsigned* MU  = (unsigned*)carve((size_t)N * H * sizeof(unsigned));
    float*    DEN = (float*)carve((size_t)N * H * sizeof(float));
    float*    EXb = (float*)carve((size_t)Etot * H * sizeof(float));

    auto layer = [&](const float* Xin, int K, const float* W, const float* asrc,
                     const float* adst, const float* bias, int Hh, int Cc,
                     int do_elu, float* Xout) {
        const int NCl = Hh * Cc;
        const int nA  = N * K;
        const int nW  = K * NCl;
        k_cvt_split<<<(nA + 255) / 256, 256, 0, stream>>>(Xin, XBh, XBl, nA);
        k_cvt_split_t<<<(nW + 255) / 256, 256, 0, stream>>>(W, WBh, WBl, K, NCl);
        const int tilesM = (N + 15) / 16, tilesN = NCl / 64;
        const int waves  = tilesM * tilesN;
        k_gemm_bf16<<<(waves + 7) / 8, 256, 0, stream>>>(XBh, XBl, WBh, WBl, HB, N, K, NCl);
        k_attn<<<(N * Hh + 255) / 256, 256, 0, stream>>>(HB, asrc, adst, ALS, ALD, N, Hh, Cc);
        k_zero_u<<<(N * Hh + 255) / 256, 256, 0, stream>>>(MU, N * Hh);
        k_zero_f<<<(N * Hh + 255) / 256, 256, 0, stream>>>(DEN, N * Hh);
        k_zero_f<<<(N * NCl + 255) / 256, 256, 0, stream>>>(AGG, N * NCl);
        k_edge_max<<<(Etot + 255) / 256, 256, 0, stream>>>(ei, ALS, ALD, MU, E, N, Hh);
        k_edge_exp<<<(Etot + 255) / 256, 256, 0, stream>>>(ei, ALS, ALD, MU, DEN, EXb, E, N, Hh);
        k_edge_scatter<<<Etot, Cc, 0, stream>>>(ei, EXb, DEN, HB, AGG, E, N, Hh, Cc);
        k_finish<<<(N * NCl + 255) / 256, 256, 0, stream>>>(AGG, bias, Xout, N * NCl, NCl, do_elu);
    };

    layer(x,  Din, W1, as1, ad1, b1, H, C,    1, XF);
    layer(XF, HC,  W2, as2, ad2, b2, H, C,    1, XF);
    layer(XF, HC,  W3, as3, ad3, b3, 1, Dout, 0, (float*)d_out);
}
